// OPAM_Small_CatPool_Module_39883066310727
// MI455X (gfx1250) — matrix operand order repaired and checked by reading, not compiled, not run on hardware
//
#include <hip/hip_runtime.h>

#define DI __device__ __forceinline__

typedef __attribute__((ext_vector_type(16))) _Float16 v16h;
typedef __attribute__((ext_vector_type(8)))  float    v8f;

// Problem constants (from reference)
constexpr int BATCH = 256;
constexpr int C     = 512;
constexpr int L     = 150;
constexpr int CI    = 128;
constexpr int KDW   = 150;     // depthwise kernel length
constexpr int LP    = 160;     // L padded to 10 tiles of 16
constexpr float EPS = 1e-5f;

// LDS strides (elements; *2B must be 16-byte aligned for wide ds access)
constexpr int VSTR = 168;      // vls  [128][LP] f16     (336 B rows)
constexpr int TSTR = 136;      // vT/qT/kT [LP][CI] f16  (272 B rows)
constexpr int XSTR = 40;       // x staging [LP][32] f16 (80 B rows)
constexpr int ESTR = 168;      // E block [32][LP] f32 (also O staging [32][128])
constexpr int PSTR = 168;      // P block [32][LP] f16

// LDS byte offsets
constexpr size_t OFF_V   = 0;
constexpr size_t SZ_V    = size_t(CI) * VSTR * 2;     // 43008
constexpr size_t OFF_VT  = OFF_V + SZ_V;
constexpr size_t SZ_T    = size_t(LP) * TSTR * 2;     // 43520
constexpr size_t OFF_QT  = OFF_VT + SZ_T;
constexpr size_t OFF_KT  = OFF_QT + SZ_T;
constexpr size_t OFF_E   = OFF_KT + SZ_T;
constexpr size_t SZ_E    = size_t(32) * ESTR * 4;     // 21504
constexpr size_t OFF_P   = OFF_E + SZ_E;
constexpr size_t SZ_P    = size_t(32) * PSTR * 2;     // 10752
constexpr size_t OFF_RED = OFF_P + SZ_P;
constexpr size_t SZ_RED  = 32 * 8 * 4;
constexpr size_t OFF_RM  = OFF_RED + SZ_RED;
constexpr size_t OFF_RS  = OFF_RM + 32 * 4;
constexpr size_t OFF_YM  = OFF_RS + 32 * 4;
constexpr size_t OFF_DW  = OFF_YM + 256 * 4;
constexpr size_t SMEM_BYTES = OFF_DW + 256 * 4;       // ~209 KB (<320 KB/WGP)

DI v8f zero8() {
  v8f z;
#pragma unroll
  for (int i = 0; i < 8; ++i) z[i] = 0.f;
  return z;
}

DI v8f wmma_f16(v16h a, v16h b, v8f c) {
  return __builtin_amdgcn_wmma_f32_16x16x32_f16(false, a, false, b, (short)0, c,
                                                false, false);
}

// ---- WMMA fragment builders (CDNA5 16x16x32 f16 layouts, wave32) ----
// A (16xK=32): lane m%16 = row; lanes<16 hold k in {0..7,16..23}, lanes>=16 +8.
// Source row-major [m][k] -> two contiguous 8-element runs per lane (merges).
DI v16h load_a_h(const _Float16* base, int stride) {
  const int lane = threadIdx.x & 31;
  const int m = lane & 15, h8 = (lane & 16) >> 1;   // 0 or 8
  v16h a;
#pragma unroll
  for (int e = 0; e < 16; ++e) {
    int k = (e & 7) + ((e >> 3) << 4) + h8;
    a[e] = base[m * stride + k];
  }
  return a;
}
DI v16h load_a_f32(const float* base, int stride) {
  const int lane = threadIdx.x & 31;
  const int m = lane & 15, h8 = (lane & 16) >> 1;
  v16h a;
#pragma unroll
  for (int e = 0; e < 16; ++e) {
    int k = (e & 7) + ((e >> 3) << 4) + h8;
    a[e] = (_Float16)base[m * stride + k];
  }
  return a;
}
// B (K=32 x 16): lane n%16 = col; lanes<16 hold k in {0..7,16..23}, lanes>=16 +8.
// Source is row-major [n][k] (i.e. B stored transposed) -> two contiguous
// 8-element runs per lane (merges into wide ds loads).
DI v16h load_bT_h(const _Float16* base, int stride) {
  const int lane = threadIdx.x & 31;
  const int n = lane & 15, k0 = lane & 16;
  v16h b;
#pragma unroll
  for (int e = 0; e < 16; ++e) b[e] = base[n * stride + ((e & 7) + ((e >> 3) << 4) + (k0 >> 1))];
  return b;
}

// ============================ Kernel 1 ============================
// One workgroup per batch element: attention + pooled/depthwise reductions
// + pointwise matvec. Emits y into d_out and pw into workspace.
__global__ __launch_bounds__(256)
void fused_attn_kernel(const float* __restrict__ x,
                       const float* __restrict__ Wv, const float* __restrict__ bv,
                       const float* __restrict__ Wq, const float* __restrict__ bq,
                       const float* __restrict__ Wk, const float* __restrict__ bk,
                       const float* __restrict__ gammap,
                       const float* __restrict__ Wd, const float* __restrict__ bd,
                       const float* __restrict__ Wp, const float* __restrict__ bp,
                       float* __restrict__ pw_ws, float* __restrict__ out) {
  extern __shared__ char smem[];
  _Float16* vls  = (_Float16*)(smem + OFF_V);    // v, [c][l]
  _Float16* vT   = (_Float16*)(smem + OFF_VT);   // v, [l][c]
  _Float16* qT   = (_Float16*)(smem + OFF_QT);   // q, [l][c]
  _Float16* kT   = (_Float16*)(smem + OFF_KT);   // k, [l][c]
  float*    Eb   = (float*)(smem + OFF_E);
  _Float16* Pb   = (_Float16*)(smem + OFF_P);
  float*    red  = (float*)(smem + OFF_RED);
  float*    rowm = (float*)(smem + OFF_RM);
  float*    rows = (float*)(smem + OFF_RS);
  float*    ymax = (float*)(smem + OFF_YM);
  float*    dws  = (float*)(smem + OFF_DW);

  const int tid  = threadIdx.x;
  const int wav  = tid >> 5;
  const int lane = tid & 31;
  const int n16  = lane & 15;
  const int hi   = lane >> 4;            // 0/1
  const int bidx = blockIdx.x;
  const float gamma = gammap[0];
  const float* xb = x + (size_t)bidx * C * L;

  // channel accumulators: running max (pool) and depthwise dot (init = bias)
  ymax[tid] = -__builtin_inff();
  dws[tid]  = bd[tid];

  // ---------- Phase 1: v = Wv x + bv -> vls [c][l] and vT [l][c] ----------
  {
    const int c0 = wav * 16;
    _Float16* xT = qT;                 // alias staging onto qT (phase-1 only)
    v8f acc[10];
#pragma unroll
    for (int it = 0; it < 10; ++it) acc[it] = zero8();
    for (int kc = 0; kc < C; kc += 32) {
      __syncthreads();                 // previous chunk's fragment reads done
      // stage x[kc..kc+31][:] transposed as f16, coalesced global reads
      for (int idx = tid; idx < 32 * L; idx += 256) {
        const int rr = idx / L, cc = idx - rr * L;
        xT[cc * XSTR + rr] = (_Float16)xb[(size_t)(kc + rr) * L + cc];
      }
      for (int idx = tid; idx < 32 * (LP - L); idx += 256) {   // zero pad cols
        const int cc = L + (idx >> 5), rr = idx & 31;
        xT[cc * XSTR + rr] = (_Float16)0.f;
      }
      __syncthreads();
      v16h a = load_a_f32(Wv + (size_t)c0 * C + kc, C);
#pragma unroll
      for (int it = 0; it < 10; ++it) {
        v16h bf = load_bT_h(xT + (it * 16) * XSTR, XSTR);
        acc[it] = wmma_f16(a, bf, acc[it]);
      }
    }
    __syncthreads();                    // last fragment reads done (qT reused)
#pragma unroll
    for (int it = 0; it < 10; ++it) {
      const int col = it * 16 + n16;
#pragma unroll
      for (int r = 0; r < 8; ++r) {
        const int row = c0 + r + 8 * hi;
        _Float16 hv = (col < L) ? (_Float16)(acc[it][r] + bv[row])
                                : (_Float16)0.f;
        vls[row * VSTR + col] = hv;
        vT[col * TSTR + row]  = hv;     // contiguous over r -> merges
      }
    }
  }
  __syncthreads();

  // ---------- Phase 2: q = Wq v + bq -> qT ; k = Wk v + bk -> kT ----------
  {
    const int c0 = wav * 16;
    v8f aq[10], ak[10];
#pragma unroll
    for (int it = 0; it < 10; ++it) { aq[it] = zero8(); ak[it] = zero8(); }
    for (int kc = 0; kc < CI; kc += 32) {
      v16h Aq = load_a_f32(Wq + (size_t)c0 * CI + kc, CI);
      v16h Ak = load_a_f32(Wk + (size_t)c0 * CI + kc, CI);
#pragma unroll
      for (int it = 0; it < 10; ++it) {
        v16h bf = load_bT_h(vT + (it * 16) * TSTR + kc, TSTR);
        aq[it] = wmma_f16(Aq, bf, aq[it]);
        ak[it] = wmma_f16(Ak, bf, ak[it]);
      }
    }
#pragma unroll
    for (int it = 0; it < 10; ++it) {
      const int col = it * 16 + n16;
#pragma unroll
      for (int r = 0; r < 8; ++r) {
        const int row = c0 + r + 8 * hi;
        qT[col * TSTR + row] = (col < L) ? (_Float16)(aq[it][r] + bq[row])
                                         : (_Float16)0.f;
        kT[col * TSTR + row] = (col < L) ? (_Float16)(ak[it][r] + bk[row])
                                         : (_Float16)0.f;
      }
    }
  }
  __syncthreads();

  // ---------- Phase 3: blocks of 32 attention rows ----------
  for (int ib = 0; ib < 5; ++ib) {
    const int i0 = ib * 32;
    // E[32][L] = q^T k  (A = qT rows i, B = kT rows j, contraction over c)
    for (int t = wav; t < 20; t += 8) {            // wave-uniform
      const int it = t / 10, jt = t % 10;
      v8f acc = zero8();
      for (int kc = 0; kc < CI; kc += 32) {
        v16h a  = load_a_h(qT + (i0 + it * 16) * TSTR + kc, TSTR);
        v16h bf = load_bT_h(kT + (jt * 16) * TSTR + kc, TSTR);
        acc = wmma_f16(a, bf, acc);
      }
#pragma unroll
      for (int r = 0; r < 8; ++r)
        Eb[(it * 16 + r + 8 * hi) * ESTR + jt * 16 + n16] = acc[r];
    }
    __syncthreads();

    // softmax over j for 32 rows (8 threads per row)
    {
      const int r = tid >> 3, sub = tid & 7;
      float m = -__builtin_inff();
      for (int j = sub; j < L; j += 8) m = fmaxf(m, Eb[r * ESTR + j]);
      red[r * 8 + sub] = m;
      __syncthreads();
      if (sub == 0) {
        float mm = red[r * 8];
#pragma unroll
        for (int u = 1; u < 8; ++u) mm = fmaxf(mm, red[r * 8 + u]);
        rowm[r] = mm;
      }
      __syncthreads();
      const float mm = rowm[r];
      float s = 0.f;
      for (int j = sub; j < L; j += 8) s += __expf(Eb[r * ESTR + j] - mm);
      red[r * 8 + sub] = s;
      __syncthreads();
      if (sub == 0) {
        float ss = 0.f;
#pragma unroll
        for (int u = 0; u < 8; ++u) ss += red[r * 8 + u];
        rows[r] = ss;
      }
      __syncthreads();
      const float inv = 1.f / rows[r];
      const bool validrow = (i0 + r) < L;
      for (int j = sub; j < LP; j += 8) {
        float p = (validrow && j < L) ? __expf(Eb[r * ESTR + j] - mm) * inv : 0.f;
        Pb[r * PSTR + j] = (_Float16)p;
      }
    }
    __syncthreads();

    // O[32][128] = P x v^T  (A = P rows i, B = vls rows c, contraction over j)
    {
      const int c0 = wav * 16;
      v8f o0 = zero8(), o1 = zero8();
      for (int kc = 0; kc < LP; kc += 32) {
        v16h bf = load_bT_h(vls + c0 * VSTR + kc, VSTR);
        v16h a0 = load_a_h(Pb + kc, PSTR);
        v16h a1 = load_a_h(Pb + 16 * PSTR + kc, PSTR);
        o0 = wmma_f16(a0, bf, o0);
        o1 = wmma_f16(a1, bf, o1);
      }
#pragma unroll
      for (int r = 0; r < 8; ++r) {
        Eb[(r + 8 * hi) * ESTR + c0 + n16]      = o0[r];
        Eb[(16 + r + 8 * hi) * ESTR + c0 + n16] = o1[r];
      }
    }
    __syncthreads();

    // fold this i-block into per-channel pool-max / depthwise-dot (att chans)
    if (tid < CI) {
      const int c = tid;
      float ym = ymax[c], dw = dws[c];
      const int imax = min(32, L - i0);
      for (int r = 0; r < imax; ++r) {
        float val = gamma * Eb[r * ESTR + c];
        ym = fmaxf(ym, val);
        dw += val * Wd[(size_t)c * KDW + (i0 + r)];
      }
      ymax[c] = ym; dws[c] = dw;
    }
    __syncthreads();
  }

  // ---------- Phase 4: v-channel reductions ----------
  if (tid < CI) {
    const int c = tid;
    float ym = ymax[CI + c], dw = dws[CI + c];
    const float* wdr = Wd + (size_t)(CI + c) * KDW;
    for (int l = 0; l < L; ++l) {
      float val = (float)vls[c * VSTR + l];
      ym = fmaxf(ym, val);
      dw += val * wdr[l];
    }
    ymax[CI + c] = ym; dws[CI + c] = dw;
  }
  __syncthreads();

  // ---------- Phase 5: pw = Wp * dw + bp ; emit y and pw ----------
  {
    float acc = bp[tid];
    const float* wpr = Wp + (size_t)tid * 256;
    for (int c2 = 0; c2 < 256; ++c2) acc += wpr[c2] * dws[c2];
    pw_ws[(size_t)bidx * 256 + tid] = acc;
    out[(size_t)bidx * 512 + 256 + tid] = ymax[tid];
  }
}

// ============================ Kernel 2 ============================
// BatchNorm2d (training stats over batch) + ReLU; one block per channel.
__global__ __launch_bounds__(256)
void bn_relu_kernel(const float* __restrict__ pw,
                    const float* __restrict__ bn_g, const float* __restrict__ bn_b,
                    float* __restrict__ out) {
  __shared__ float s1[256], s2[256];
  const int o = blockIdx.x, t = threadIdx.x;
  const float v = pw[(size_t)t * 256 + o];
  s1[t] = v; s2[t] = v * v;
  __syncthreads();
  for (int st = 128; st > 0; st >>= 1) {
    if (t < st) { s1[t] += s1[t + st]; s2[t] += s2[t + st]; }
    __syncthreads();
  }
  const float mean = s1[0] * (1.f / 256.f);
  const float var  = s2[0] * (1.f / 256.f) - mean * mean;
  const float sc   = rsqrtf(var + EPS) * bn_g[o];
  const float res  = (v - mean) * sc + bn_b[o];
  out[(size_t)t * 512 + o] = fmaxf(res, 0.f);
}

extern "C" void kernel_launch(void* const* d_in, const int* in_sizes, int n_in,
                              void* d_out, int out_size, void* d_ws, size_t ws_size,
                              hipStream_t stream) {
  const float* x   = (const float*)d_in[0];
  const float* Wv  = (const float*)d_in[1];
  const float* bv  = (const float*)d_in[2];
  const float* Wq  = (const float*)d_in[3];
  const float* bq  = (const float*)d_in[4];
  const float* Wk  = (const float*)d_in[5];
  const float* bk  = (const float*)d_in[6];
  const float* gm  = (const float*)d_in[7];
  const float* Wd  = (const float*)d_in[8];
  const float* bd  = (const float*)d_in[9];
  const float* Wp  = (const float*)d_in[10];
  const float* bp  = (const float*)d_in[11];
  const float* bng = (const float*)d_in[12];
  const float* bnb = (const float*)d_in[13];
  float* out   = (float*)d_out;
  float* pw_ws = (float*)d_ws;   // 256*256 f32 = 256 KB scratch

  // Dynamic LDS > default 64 KB: raise the opt-in limit (idempotent, capture-safe)
  (void)hipFuncSetAttribute(reinterpret_cast<const void*>(&fused_attn_kernel),
                            hipFuncAttributeMaxDynamicSharedMemorySize,
                            (int)SMEM_BYTES);

  fused_attn_kernel<<<BATCH, 256, SMEM_BYTES, stream>>>(
      x, Wv, bv, Wq, bq, Wk, bk, gm, Wd, bd, Wp, bp, pw_ws, out);
  bn_relu_kernel<<<256, 256, 0, stream>>>(pw_ws, bng, bnb, out);
}
